// TemporalCrosscoder_65128884076998
// MI455X (gfx1250) — matrix operand order repaired and checked by reading, not compiled, not run on hardware
//
#include <hip/hip_runtime.h>

// Problem constants
#define D_IN   768
#define TSTEPS 4
#define KDIM   3072      // T * D_IN (GEMM K)
#define HDIM   32768     // D_SAE   (GEMM N)
#define BATCH  4096      // B       (GEMM M)
#define TOPK   32

typedef _Float16 half_t;
typedef __attribute__((ext_vector_type(8)))  _Float16 v8h;
typedef __attribute__((ext_vector_type(16))) _Float16 v16h;
typedef __attribute__((ext_vector_type(8)))  float    v8f;

#if __has_builtin(__builtin_amdgcn_sched_group_barrier)
#define SCHED_GROUP(mask, size) __builtin_amdgcn_sched_group_barrier((mask), (size), 0)
#else
#define SCHED_GROUP(mask, size)
#endif

// ---------------------------------------------------------------------------
// 1) Convert W_enc (H x K, f32) -> f16
// ---------------------------------------------------------------------------
__global__ void cvt_wenc_f16(const float* __restrict__ w, half_t* __restrict__ o) {
    size_t i = ((size_t)blockIdx.x * blockDim.x + threadIdx.x) * 4;
    float4 f = *(const float4*)(w + i);
    o[i + 0] = (half_t)f.x;
    o[i + 1] = (half_t)f.y;
    o[i + 2] = (half_t)f.z;
    o[i + 3] = (half_t)f.w;
}

// ---------------------------------------------------------------------------
// 2) x_flat = (x - b_dec) -> f16   (x is (B,T,D) row-major == (B, K))
// ---------------------------------------------------------------------------
__global__ void cvt_x_center_f16(const float* __restrict__ x,
                                 const float* __restrict__ b_dec,
                                 half_t* __restrict__ o) {
    size_t i = ((size_t)blockIdx.x * blockDim.x + threadIdx.x) * 4;
    int c = (int)(i % KDIM);                   // KDIM % 4 == 0 -> no row crossing
    float4 f  = *(const float4*)(x + i);
    float4 bd = *(const float4*)(b_dec + c);
    o[i + 0] = (half_t)(f.x - bd.x);
    o[i + 1] = (half_t)(f.y - bd.y);
    o[i + 2] = (half_t)(f.z - bd.z);
    o[i + 3] = (half_t)(f.w - bd.w);
}

// ---------------------------------------------------------------------------
// 3) Tiled transpose W_dec (K x H) -> W_decT (H x K)  [f32]
// ---------------------------------------------------------------------------
__global__ void transpose_wdec(const float* __restrict__ wdec,
                               float* __restrict__ wdecT) {
    __shared__ float tile[32][33];
    const int h0 = blockIdx.x * 32;
    const int c0 = blockIdx.y * 32;
    const int tx = threadIdx.x & 31;
    const int ty = threadIdx.x >> 5;           // 0..7
#pragma unroll
    for (int i = 0; i < 4; ++i) {
        int c = ty + i * 8;
        tile[c][tx] = wdec[(size_t)(c0 + c) * HDIM + (h0 + tx)];   // coalesced in h
    }
    __syncthreads();
#pragma unroll
    for (int i = 0; i < 4; ++i) {
        int h = ty + i * 8;
        wdecT[(size_t)(h0 + h) * KDIM + (c0 + tx)] = tile[tx][h];  // coalesced in c
    }
}

// ---------------------------------------------------------------------------
// Async 16B global -> LDS copy (CDNA5 GLOBAL_LOAD_ASYNC_TO_LDS, ASYNCcnt)
// ---------------------------------------------------------------------------
__device__ __forceinline__ void async_cp16(void* lds_dst, const void* gsrc) {
    asm volatile("global_load_async_to_lds_b128 %0, %1, off"
                 :: "v"((unsigned)(unsigned long long)lds_dst),   // low 32b of LDS generic addr
                    "v"((unsigned long long)gsrc)
                 : "memory");
}

// ---------------------------------------------------------------------------
// 4) Encoder GEMM: pre[M,N] = A[M,K](f16) * W_enc[N,K](f16)^T + b_enc
//    Block tile 64(M) x 256(N), BK=32, double-buffered LDS filled by
//    async-to-LDS copies. 8 waves: 4(M) x 2(N); wave = 16x128.
//    Scheduler shaped: {1 WMMA | 2 DS-reads} to keep <=2 fragments in flight
//    (avoids acc spills -> v_mov/v_nop WMMA hazards seen otherwise).
// ---------------------------------------------------------------------------
__global__ void __launch_bounds__(256)
encode_gemm_wmma(const half_t* __restrict__ A,      // (BATCH, KDIM)
                 const half_t* __restrict__ Bw,     // (HDIM, KDIM) = W_enc rows
                 const float*  __restrict__ b_enc,  // (HDIM)
                 float* __restrict__ pre) {         // (BATCH, HDIM)
    __shared__ __align__(32) half_t As[2][64 * 32];    //  2 x 4 KB
    __shared__ __align__(32) half_t Bs[2][256 * 32];   //  2 x 16 KB

    const int tid    = threadIdx.x;
    const int lane   = tid & 31;
    const int waveId = tid >> 5;
    const int waveM  = waveId & 3;      // 0..3
    const int waveN  = waveId >> 2;     // 0..1
    const int lr     = lane & 15;
    const int lh     = lane >> 4;
    // M fastest-varying across blocks: concurrent blocks share the same W_enc
    // slice + the whole activation matrix -> L2-resident re-reads.
    const int rowBase = blockIdx.x * 64;
    const int colBase = blockIdx.y * 256;

    const v8f zero = {0.f, 0.f, 0.f, 0.f, 0.f, 0.f, 0.f, 0.f};
    v8f acc[8];
#pragma unroll
    for (int j = 0; j < 8; ++j) acc[j] = zero;

    const int aRow = tid >> 2, aChunk = tid & 3;   // A: 256 x 16B chunks

    // Issue one K-tile of async copies into buffer p (5 x 16B per thread)
    auto issue_tile = [&](int p, int kb) {
        async_cp16(&As[p][aRow * 32 + aChunk * 8],
                   &A[(size_t)(rowBase + aRow) * KDIM + kb + aChunk * 8]);
#pragma unroll
        for (int i = 0; i < 4; ++i) {
            int cid = tid + i * 256;
            int n = cid >> 2, c = cid & 3;
            async_cp16(&Bs[p][n * 32 + c * 8],
                       &Bw[(size_t)(colBase + n) * KDIM + kb + c * 8]);
        }
    };

    const int NTILES = KDIM / 32;       // 96
    issue_tile(0, 0);

    for (int i = 0; i < NTILES; ++i) {
        const int p = i & 1;
        __syncthreads();                       // everyone done reading buf p^1
        if (i + 1 < NTILES) {
            issue_tile(p ^ 1, (i + 1) * 32);   // prefetch next tile
            asm volatile("s_wait_asynccnt 0x5" ::: "memory");  // own buf-p copies done
        } else {
            asm volatile("s_wait_asynccnt 0x0" ::: "memory");
        }
        __syncthreads();                       // all waves' buf-p copies done

        // A fragment (16x32): lanes 0-15 row M=lr K 0..15; lanes 16-31 K 16..31
        const int ar = waveM * 16 + lr;
        v8h alo = *(const v8h*)&As[p][ar * 32 + lh * 16];
        v8h ahi = *(const v8h*)&As[p][ar * 32 + lh * 16 + 8];
        v16h afrag = __builtin_shufflevector(alo, ahi,
                       0,1,2,3,4,5,6,7,8,9,10,11,12,13,14,15);
#pragma unroll
        for (int j = 0; j < 8; ++j) {
            // B fragment (32x16): lanes 0-15 -> N=lr, K 0..15; lanes 16-31 -> K 16..31
            const int bn = waveN * 128 + j * 16 + lr;
            v16h bfrag = *(const v16h*)&Bs[p][bn * 32 + lh * 16];
            acc[j] = __builtin_amdgcn_wmma_f32_16x16x32_f16(
                         false, afrag, false, bfrag, (short)0, acc[j], false, false);
        }
        // Pipeline shape: A(2)+B0(2) ds-reads, then alternate WMMA / next-B reads.
        SCHED_GROUP(0x100, 4);                 // 4 DS reads (A frag + B frag 0)
#pragma unroll
        for (int j = 0; j < 7; ++j) {
            SCHED_GROUP(0x008, 1);             // 1 WMMA
            SCHED_GROUP(0x100, 2);             // 2 DS reads (B frag j+1)
        }
        SCHED_GROUP(0x008, 1);                 // final WMMA
    }

    // Epilogue: C layout -> VGPR r: lanes 0-15 M=r, lanes 16-31 M=r+8; N = lr
#pragma unroll
    for (int j = 0; j < 8; ++j) {
        const int col = colBase + waveN * 128 + j * 16 + lr;
        const float be = b_enc[col];
#pragma unroll
        for (int r = 0; r < 8; ++r) {
            const int row = rowBase + waveM * 16 + lh * 8 + r;
            pre[(size_t)row * HDIM + col] = acc[j][r] + be;
        }
    }
}

// ---------------------------------------------------------------------------
// 5) Top-k(32) per row; in-place: reads pre from u-region, writes k-sparse u.
//    Full row staged in LDS (128KB; fine within 320KB/WGP).
// ---------------------------------------------------------------------------
__global__ void __launch_bounds__(256)
topk_scatter(float* __restrict__ u,            // (BATCH, HDIM), holds pre on entry
             float* __restrict__ valsW,        // (BATCH, TOPK)
             int*   __restrict__ idxW) {       // (BATCH, TOPK)
    extern __shared__ char smem[];
    float* sv   = (float*)smem;                // HDIM
    float* rv   = sv + HDIM;                   // 256
    float* selV = rv + 256;                    // 32
    int*   ri   = (int*)(selV + 32);           // 256
    int*   selI = ri + 256;                    // 32

    const int row = blockIdx.x, tid = threadIdx.x;
    float* preRow = u + (size_t)row * HDIM;

#pragma unroll 4
    for (int i = 0; i < 32; ++i) {             // 8192 float4s / 256 threads
        int e4 = tid + i * 256;
        float4 f = *(const float4*)(preRow + (size_t)e4 * 4);
        int e = e4 * 4;
        sv[e] = f.x; sv[e + 1] = f.y; sv[e + 2] = f.z; sv[e + 3] = f.w;
    }
    __syncthreads();

    for (int it = 0; it < TOPK; ++it) {
        // local argmax over contiguous segment (lowest-index tie-break)
        float bv = -__builtin_inff();
        int   bi = tid * 128;
        for (int i = 0; i < 128; ++i) {
            int e = tid * 128 + i;
            float v = sv[e];
            if (v > bv) { bv = v; bi = e; }
        }
        rv[tid] = bv; ri[tid] = bi;
        __syncthreads();
        for (int s = 128; s > 0; s >>= 1) {
            if (tid < s) {
                float v2 = rv[tid + s]; int i2 = ri[tid + s];
                if (v2 > rv[tid] || (v2 == rv[tid] && i2 < ri[tid])) {
                    rv[tid] = v2; ri[tid] = i2;
                }
            }
            __syncthreads();
        }
        if (tid == 0) { selV[it] = rv[0]; selI[it] = ri[0]; sv[ri[0]] = -__builtin_inff(); }
        __syncthreads();
    }

    // zero the u row, then scatter relu(vals)
    const float4 z = {0.f, 0.f, 0.f, 0.f};
#pragma unroll 4
    for (int i = 0; i < 32; ++i) {
        int e4 = tid + i * 256;
        *(float4*)(preRow + (size_t)e4 * 4) = z;
    }
    __syncthreads();
    if (tid < TOPK) {
        float v = selV[tid]; v = v > 0.f ? v : 0.f;
        int h = selI[tid];
        preRow[h] = v;
        valsW[row * TOPK + tid] = v;
        idxW[row * TOPK + tid]  = h;
    }
}

// ---------------------------------------------------------------------------
// 6) Decode + per-(b,t) loss partial
// ---------------------------------------------------------------------------
__global__ void __launch_bounds__(256)
decode_loss(const float* __restrict__ x,
            const float* __restrict__ b_dec,
            const float* __restrict__ wdecT,   // (HDIM, KDIM)
            const float* __restrict__ valsW,
            const int*   __restrict__ idxW,
            float* __restrict__ xhat,          // (BATCH*TSTEPS, D_IN)
            float* __restrict__ partials) {    // (BATCH*TSTEPS)
    __shared__ float sv[TOPK];
    __shared__ int   si[TOPK];
    __shared__ float red[256];

    const int blk = blockIdx.x;            // b*4 + t
    const int b = blk >> 2, t = blk & 3;
    const int tid = threadIdx.x;
    if (tid < TOPK) { sv[tid] = valsW[b * TOPK + tid]; si[tid] = idxW[b * TOPK + tid]; }
    __syncthreads();

    const int base = t * D_IN;
    float acc[3];
#pragma unroll
    for (int i = 0; i < 3; ++i) acc[i] = b_dec[base + tid + i * 256];

#pragma unroll 8
    for (int j = 0; j < TOPK; ++j) {
        const float v = sv[j];
        const size_t hb = (size_t)si[j] * KDIM + base;
#pragma unroll
        for (int i = 0; i < 3; ++i)
            acc[i] += v * wdecT[hb + tid + i * 256];   // coalesced across threads
    }

    float local = 0.f;
#pragma unroll
    for (int i = 0; i < 3; ++i) {
        const size_t o = (size_t)blk * D_IN + tid + i * 256;
        xhat[o] = acc[i];
        float d = acc[i] - x[o];
        local += d * d;
    }
    red[tid] = local;
    __syncthreads();
    for (int s = 128; s > 0; s >>= 1) {
        if (tid < s) red[tid] += red[tid + s];
        __syncthreads();
    }
    if (tid == 0) partials[blk] = red[0];
}

// ---------------------------------------------------------------------------
// 7) Deterministic final loss reduction: mean over B*T
// ---------------------------------------------------------------------------
__global__ void reduce_loss(const float* __restrict__ partials, float* __restrict__ out0) {
    __shared__ float red[256];
    const int tid = threadIdx.x;
    float s = 0.f;
#pragma unroll
    for (int i = 0; i < 64; ++i) s += partials[tid + i * 256];
    red[tid] = s;
    __syncthreads();
    for (int k = 128; k > 0; k >>= 1) {
        if (tid < k) red[tid] += red[tid + k];
        __syncthreads();
    }
    if (tid == 0) out0[0] = red[0] * (1.0f / (float)(BATCH * TSTEPS));
}

// ---------------------------------------------------------------------------
extern "C" void kernel_launch(void* const* d_in, const int* in_sizes, int n_in,
                              void* d_out, int out_size, void* d_ws, size_t ws_size,
                              hipStream_t stream) {
    const float* x     = (const float*)d_in[0];   // (B,T,D)
    const float* W_enc = (const float*)d_in[1];   // (H, K)
    const float* b_enc = (const float*)d_in[2];   // (H)
    const float* W_dec = (const float*)d_in[3];   // (K, H) flattened from (T,D,H)
    const float* b_dec = (const float*)d_in[4];   // (K)
    // d_in[5] = k (always 32 here)

    float* out = (float*)d_out;
    float* loss_out = out;                                  // [0]
    float* xhat_out = out + 1;                              // (B*T*D)
    float* u_out    = out + 1 + (size_t)BATCH * KDIM;       // (B, H) — also pre scratch

    // Workspace carve
    char* w = (char*)d_ws;
    half_t* wencH = (half_t*)w;  w += (size_t)HDIM * KDIM * sizeof(half_t);   // 201.3 MB
    half_t* xH    = (half_t*)w;  w += (size_t)BATCH * KDIM * sizeof(half_t);  //  25.2 MB
    float*  wdecT = (float*)w;   w += (size_t)HDIM * KDIM * sizeof(float);    // 402.7 MB
    float*  valsW = (float*)w;   w += (size_t)BATCH * TOPK * sizeof(float);
    int*    idxW  = (int*)w;     w += (size_t)BATCH * TOPK * sizeof(int);
    float*  parts = (float*)w;   w += (size_t)BATCH * TSTEPS * sizeof(float);

    // 1-2) f16 conversions
    cvt_wenc_f16<<<(unsigned)((size_t)HDIM * KDIM / 4 / 256), 256, 0, stream>>>(W_enc, wencH);
    cvt_x_center_f16<<<(unsigned)((size_t)BATCH * KDIM / 4 / 256), 256, 0, stream>>>(x, b_dec, xH);
    // 3) transpose W_dec for coalesced decode gathers
    transpose_wdec<<<dim3(HDIM / 32, KDIM / 32), 256, 0, stream>>>(W_dec, wdecT);
    // 4) WMMA encoder GEMM -> pre (stored in u region of d_out); M fastest for L2 reuse
    encode_gemm_wmma<<<dim3(BATCH / 64, HDIM / 256), 256, 0, stream>>>(xH, wencH, b_enc, u_out);
    // 5) top-k + relu + scatter (in-place pre -> u)
    size_t topkShmem = (size_t)(HDIM + 256 + 32) * sizeof(float) + (256 + 32) * sizeof(int);
    topk_scatter<<<BATCH, 256, topkShmem, stream>>>(u_out, valsW, idxW);
    // 6) sparse decode + loss partials
    decode_loss<<<BATCH * TSTEPS, 256, 0, stream>>>(x, b_dec, wdecT, valsW, idxW, xhat_out, parts);
    // 7) final deterministic reduction
    reduce_loss<<<1, 256, 0, stream>>>(parts, loss_out);
}
